// BaseRNN_56762287784464
// MI455X (gfx1250) — matrix operand order repaired and checked by reading, not compiled, not run on hardware
//
#include <hip/hip_runtime.h>
#include <math.h>

// Problem constants (match reference).
constexpr int kB  = 256;   // batch
constexpr int kT  = 2048;  // seq len
constexpr int kH  = 128;   // hidden == embedding dim
constexpr int kBT = 16;    // batch rows per block (WMMA M)
constexpr int kNT = 8;     // N tiles of 16 -> H=128, one per wave
constexpr int kTHREADS = kNT * 32;  // 8 wave32 waves
constexpr int kPF = 8;     // embedding prefetch distance (steps)

typedef __attribute__((ext_vector_type(16))) __bf16 v16bf;
typedef __attribute__((ext_vector_type(8)))  __bf16 v8bf;
typedef __attribute__((ext_vector_type(8)))  float  v8f;
typedef __attribute__((ext_vector_type(4)))  float  v4f;

union FragU { v16bf v; v8bf h[2]; __bf16 e[16]; };

struct __align__(16) Smem {
    float  h1f[kBT * kH];       // layer-1 hidden f32, classifier handoff only
    __bf16 xbf[2][kBT * kH];    // double-buffered embedded token tile, bf16
    __bf16 h0b[kBT * kH];       // layer-0 hidden, bf16 (WMMA A operand)
    __bf16 h1b[kBT * kH];       // layer-1 hidden, bf16 (WMMA A operand)
};

// Branchless tanh from HW transcendentals: v_exp_f32 (2^x) + v_rcp_f32.
__device__ __forceinline__ float fast_tanh(float x) {
    const float t = __builtin_amdgcn_exp2f(__builtin_fabsf(x) * -2.885390082f);
    const float r = (1.f - t) * __builtin_amdgcn_rcpf(1.f + t);
    return __builtin_copysignf(r, x);
}

__device__ __forceinline__ float fast_sigmoid(float z) {
    return __builtin_amdgcn_rcpf(1.f + __builtin_amdgcn_exp2f(z * -1.442695041f));
}

// A-fragment (16x32 bf16) per ISA 16-bit A-matrix layout:
// lane L holds row M=L&15; e=0..7 -> K = kc*32 + (L>=16?8:0) + e,
// e=8..15 -> K = kc*32 + 16 + (L>=16?8:0) + (e-8).  Both halves are
// 8 contiguous bf16 in a row-major [16][128] buffer -> two ds_load_b128.
__device__ __forceinline__ v16bf ldA(const __bf16* act, int lane, int kc) {
    const int row = lane & 15;
    const int hi8 = (lane >> 4) << 3;
    FragU u;
    u.h[0] = *(const v8bf*)(act + row * kH + kc * 32 + hi8);
    u.h[1] = *(const v8bf*)(act + row * kH + kc * 32 + 16 + hi8);
    return u.v;
}

// B-fragment (32x16 bf16) gathered straight from the global f32 weight
// (row-major [K][N]) into WMMA layout: lane L holds column N=L&15,
// e=0..7 -> K = kc*32 + (L>=16?8:0) + e, e=8..15 -> K = kc*32 + 16 + (L>=16?8:0) + (e-8).  Done once at kernel start.
__device__ __forceinline__ v16bf ldBglobal(const float* w, int lane, int ntile, int kc) {
    const int n  = ntile * 16 + (lane & 15);
    const int kb = kc * 32 + ((lane >> 4) << 3);
    FragU u;
#pragma unroll
    for (int e = 0; e < 16; ++e) u.e[e] = (__bf16)w[(kb + e + ((e >> 3) << 3)) * kH + n];
    return u.v;
}

// One RNN layer, one timestep:  h = (t<len) ? tanh(in@Wih + h@Whh + b) : h
// f32 state in hreg[8] (thread-private); bf16 copy republished to LDS.
// Ends *before* the publish barrier (caller supplies it).
__device__ __forceinline__ void layer_step(
    const __bf16* __restrict__ inb, const v16bf* Bih, const v16bf* Bhh,
    float* hreg, __bf16* __restrict__ hb,
    float bs, const int* lenv, int t, int lane, int ntile)
{
    v16bf Ax[4], Ah[4];
#pragma unroll
    for (int kc = 0; kc < 4; ++kc) Ax[kc] = ldA(inb, lane, kc);
#pragma unroll
    for (int kc = 0; kc < 4; ++kc) Ah[kc] = ldA(hb, lane, kc);

    // Two independent accumulator chains (depth 4 each) instead of one
    // depth-8 chain; merged below with one v_add_f32 per element.
    v8f acc0 = {}, acc1 = {};
#pragma unroll
    for (int kc = 0; kc < 4; ++kc) {
        acc0 = __builtin_amdgcn_wmma_f32_16x16x32_bf16(
            false, Ax[kc], false, Bih[kc], (short)0, acc0, false, false);
        acc1 = __builtin_amdgcn_wmma_f32_16x16x32_bf16(
            false, Ah[kc], false, Bhh[kc], (short)0, acc1, false, false);
    }

    __syncthreads();  // all waves done reading hb before anyone rewrites it

    // C/D layout: lane 0-15 -> N=lane, VGPR g -> M=g; lane 16-31 -> M=g+8.
    const int n  = ntile * 16 + (lane & 15);
    const int mb = (lane >> 4) << 3;
#pragma unroll
    for (int g = 0; g < 8; ++g) {
        const float hnew = fast_tanh(acc0[g] + acc1[g] + bs);
        const float hv   = (t < lenv[g]) ? hnew : hreg[g];  // pure v_cndmask
        hreg[g] = hv;
        hb[(mb + g) * kH + n] = (__bf16)hv;
    }
}

__global__ __launch_bounds__(kTHREADS)
void rnn_fused_kernel(const int* __restrict__ x, const int* __restrict__ lengths,
                      const float* __restrict__ emb,
                      const float* __restrict__ W_ih, const float* __restrict__ W_hh,
                      const float* __restrict__ bias, const float* __restrict__ cls_w,
                      const float* __restrict__ cls_b, float* __restrict__ out)
{
    __shared__ Smem s;
    const int tid   = threadIdx.x;
    const int lane  = tid & 31;
    const int ntile = tid >> 5;       // wave id == N tile
    const int b0    = blockIdx.x * kBT;

    // Weights: bf16 B-fragments resident in registers for the whole sequence.
    v16bf Bih0[4], Bhh0[4], Bih1[4], Bhh1[4];
#pragma unroll
    for (int kc = 0; kc < 4; ++kc) {
        Bih0[kc] = ldBglobal(W_ih,            lane, ntile, kc);
        Bhh0[kc] = ldBglobal(W_hh,            lane, ntile, kc);
        Bih1[kc] = ldBglobal(W_ih + kH * kH,  lane, ntile, kc);
        Bhh1[kc] = ldBglobal(W_hh + kH * kH,  lane, ntile, kc);
    }

    // Per-thread constants: column bias, lengths of my 8 C-tile rows.
    const int n  = ntile * 16 + (lane & 15);
    const int mb = (lane >> 4) << 3;
    const float bs0 = bias[n];
    const float bs1 = bias[kH + n];
    int lenv[8];
#pragma unroll
    for (int g = 0; g < 8; ++g) lenv[g] = lengths[b0 + mb + g];

    // f32 hidden state lives in registers (same thread owns same elements).
    float h0r[8], h1r[8];
#pragma unroll
    for (int g = 0; g < 8; ++g) { h0r[g] = 0.f; h1r[g] = 0.f; }

    for (int i = tid; i < kBT * kH; i += kTHREADS) {
        s.h0b[i] = (__bf16)0.f; s.h1b[i] = (__bf16)0.f;
    }

    // Embedding gather geometry: 16 threads per batch row, 8 floats each.
    const int  row  = tid >> 4;
    const int  cb   = (tid & 15) * 8;
    const long xrow = (long)(b0 + row) * kT;

    // Prologue: stage x tile for t=0.
    {
        const float* src = emb + (long)x[xrow] * kH + cb;
        const v4f a  = *(const v4f*)src;
        const v4f b4 = *(const v4f*)(src + 4);
        __bf16* d = s.xbf[0] + row * kH + cb;
#pragma unroll
        for (int i = 0; i < 4; ++i) { d[i] = (__bf16)a[i]; d[4 + i] = (__bf16)b4[i]; }
    }
    __syncthreads();

    for (int t = 0; t < kT; ++t) {
        const int cur = t & 1;

        // Issue next step's embedding loads; HBM latency hides behind the
        // 16 WMMAs + tanh below.  Deep prefetch kPF steps ahead.
        v4f a = {}, b4 = {};
        if (t + 1 < kT) {
            const float* src = emb + (long)x[xrow + t + 1] * kH + cb;
            a  = *(const v4f*)src;
            b4 = *(const v4f*)(src + 4);
        }
        if (((tid & 15) == 0) && (t + kPF < kT))
            __builtin_prefetch(emb + (long)x[xrow + t + kPF] * kH, 0, 1);

        layer_step(s.xbf[cur], Bih0, Bhh0, h0r, s.h0b, bs0, lenv, t, lane, ntile);
        __syncthreads();  // publish h0b for layer 1
        layer_step(s.h0b,      Bih1, Bhh1, h1r, s.h1b, bs1, lenv, t, lane, ntile);

        if (t + 1 < kT) {  // uniform condition
            __bf16* d = s.xbf[cur ^ 1] + row * kH + cb;
#pragma unroll
            for (int i = 0; i < 4; ++i) { d[i] = (__bf16)a[i]; d[4 + i] = (__bf16)b4[i]; }
        }
        __syncthreads();  // publish h1b + next x tile together
    }

    // Hand the register-resident layer-1 f32 state to LDS once.
#pragma unroll
    for (int g = 0; g < 8; ++g) s.h1f[(mb + g) * kH + n] = h1r[g];
    __syncthreads();

    // Classifier head: sigmoid(h1 @ cls_w + cls_b), one row per thread.
    if (tid < kBT) {
        float z = cls_b[0];
        for (int k = 0; k < kH; ++k) z += s.h1f[tid * kH + k] * cls_w[k];
        out[b0 + tid] = fast_sigmoid(z);
    }
}

extern "C" void kernel_launch(void* const* d_in, const int* in_sizes, int n_in,
                              void* d_out, int out_size, void* d_ws, size_t ws_size,
                              hipStream_t stream) {
    (void)in_sizes; (void)n_in; (void)out_size; (void)d_ws; (void)ws_size;
    const int*   x       = (const int*)d_in[0];
    const int*   lengths = (const int*)d_in[1];
    const float* emb     = (const float*)d_in[2];
    const float* W_ih    = (const float*)d_in[3];
    const float* W_hh    = (const float*)d_in[4];
    const float* bias    = (const float*)d_in[5];
    const float* cls_w   = (const float*)d_in[6];
    const float* cls_b   = (const float*)d_in[7];
    float*       out     = (float*)d_out;

    dim3 grid(kB / kBT);      // 16 blocks, one 16-row batch tile each
    dim3 block(kTHREADS);     // 8 wave32 waves
    rnn_fused_kernel<<<grid, block, 0, stream>>>(
        x, lengths, emb, W_ih, W_hh, bias, cls_w, cls_b, out);
}
